// EfficientBalancedIPRMPNNModel_89876485636288
// MI455X (gfx1250) — matrix operand order repaired and checked by reading, not compiled, not run on hardware
//
#include <hip/hip_runtime.h>
#include <stdint.h>

typedef __attribute__((ext_vector_type(16))) __bf16 v16bf;
typedef __attribute__((ext_vector_type(8)))  float  v8f;

#define TOTALN 32768
#define NBAT   64
#define NNODE  512
#define NVIRT  64
#define NHID   256
#define NEDGE  1048576

__device__ __forceinline__ unsigned short f2bf(float f) {
  unsigned int u = __float_as_uint(f);
  u += 0x7FFFu + ((u >> 16) & 1u);          // round-to-nearest-even
  return (unsigned short)(u >> 16);
}

union Frag { uint4 q[2]; v16bf v; };

// ---------------------------------------------------------------------------
// General GEMM: C[M,N] = A[M,K] @ W[K,N] (+bias) (+relu), fp32 in/out,
// bf16 WMMA 16x16x32 inner product. Requires M%128==0, K%32==0, N%128==0.
// Block tile 128x128, 256 threads = 8 waves in 4(M) x 2(N), wave = 32x64.
// ---------------------------------------------------------------------------
__global__ __launch_bounds__(256) void gemm_bf16(
    const float* __restrict__ A, const float* __restrict__ W,
    const float* __restrict__ bias, float* __restrict__ C,
    int M, int K, int N, int relu)
{
  __shared__ __align__(16) unsigned short As[128][40];  // [row][k], pad->80B rows
  __shared__ __align__(16) unsigned short Bs[128][40];  // [col][k] (transposed W)

  const int tid  = threadIdx.x;
  const int mb   = blockIdx.x * 128;
  const int nb   = blockIdx.y * 128;
  const int wave = tid >> 5, lane = tid & 31;
  const int waveM = wave & 3, waveN = wave >> 2;
  const int r = lane & 15, kg = lane >> 4;

  v8f acc[2][4] = {};

  for (int kb = 0; kb < K; kb += 32) {
    // ---- stage A tile 128x32 (fp32 -> bf16) ----
    for (int i = 0; i < 4; i++) {
      int idx = tid + i * 256;
      int row = idx >> 3, c4 = (idx & 7) << 2;
      float4 v = *(const float4*)&A[(size_t)(mb + row) * K + kb + c4];
      uint2 p;
      p.x = (unsigned)f2bf(v.x) | ((unsigned)f2bf(v.y) << 16);
      p.y = (unsigned)f2bf(v.z) | ((unsigned)f2bf(v.w) << 16);
      *(uint2*)&As[row][c4] = p;
    }
    // ---- stage B tile 32x128, transposed into Bs[col][k] ----
    for (int i = 0; i < 4; i++) {
      int idx = tid + i * 256;
      int kr = idx >> 5, c4 = (idx & 31) << 2;
      float4 v = *(const float4*)&W[(size_t)(kb + kr) * N + nb + c4];
      Bs[c4 + 0][kr] = f2bf(v.x);
      Bs[c4 + 1][kr] = f2bf(v.y);
      Bs[c4 + 2][kr] = f2bf(v.z);
      Bs[c4 + 3][kr] = f2bf(v.w);
    }
    __syncthreads();

    Frag a[2], b[4];
    for (int i = 0; i < 2; i++) {
      const unsigned short* p = &As[waveM * 32 + i * 16 + r][0];
      a[i].q[0] = *(const uint4*)(p + kg * 16);       // K = kg*16 + 0..7
      a[i].q[1] = *(const uint4*)(p + kg * 16 + 8);   // K = kg*16 + 8..15
    }
    for (int j = 0; j < 4; j++) {
      const unsigned short* p = &Bs[waveN * 64 + j * 16 + r][kg * 16];
      b[j].q[0] = *(const uint4*)(p);                 // K = kg*16 + 0..7
      b[j].q[1] = *(const uint4*)(p + 8);             // K = kg*16 + 8..15
    }
    for (int i = 0; i < 2; i++)
      for (int j = 0; j < 4; j++)
        acc[i][j] = __builtin_amdgcn_wmma_f32_16x16x32_bf16(
            false, a[i].v, false, b[j].v, (short)0, acc[i][j], false, false);
    __syncthreads();
  }

  // ---- epilogue: bias + relu, C/D layout: lane n = lane&15, row = (lane>>4)*8+v
  for (int j = 0; j < 4; j++) {
    int gcol = nb + waveN * 64 + j * 16 + r;
    float bv = bias ? bias[gcol] : 0.0f;
    for (int i = 0; i < 2; i++) {
      int rowbase = mb + waveM * 32 + i * 16 + kg * 8;
      for (int v = 0; v < 8; v++) {
        float val = acc[i][j][v] + bv;
        if (relu) val = fmaxf(val, 0.0f);
        C[(size_t)(rowbase + v) * N + gcol] = val;
      }
    }
  }
}

// ---------------------------------------------------------------------------
// vn[b] = mw[b]^T (V x N) @ g[b] (N x H), with mw[b][n][v] = ew[b][n][v]*scl[b][n]
// Block tile: M=64 (all V), N=128; waves 4(M) x 2(N), each 16x64.
// ---------------------------------------------------------------------------
__global__ __launch_bounds__(256) void vn_gemm(
    const float* __restrict__ ew, const float* __restrict__ scl,
    const float* __restrict__ g, float* __restrict__ vn)
{
  __shared__ __align__(16) unsigned short As[64][40];   // [v][k]
  __shared__ __align__(16) unsigned short Bs[128][40];  // [h][k]
  const int tid = threadIdx.x;
  const int b = blockIdx.x;
  const int nb = blockIdx.y * 128;
  const int wave = tid >> 5, lane = tid & 31;
  const int waveM = wave & 3, waveN = wave >> 2;
  const int r = lane & 15, kg = lane >> 4;
  const float* ewb  = ew  + (size_t)b * NNODE * NVIRT;
  const float* sclb = scl + (size_t)b * NNODE;
  const float* gb   = g   + (size_t)b * NNODE * NHID;

  v8f acc[4] = {};

  for (int kb = 0; kb < NNODE; kb += 32) {
    for (int i = 0; i < 8; i++) {          // A: 64 v x 32 k (k = node index)
      int idx = tid + i * 256;
      int kc = idx >> 6, v = idx & 63;
      As[v][kc] = f2bf(ewb[(size_t)(kb + kc) * NVIRT + v] * sclb[kb + kc]);
    }
    for (int i = 0; i < 4; i++) {          // B: 32 k x 128 h -> Bs[h][k]
      int idx = tid + i * 256;
      int kr = idx >> 5, c4 = (idx & 31) << 2;
      float4 v = *(const float4*)&gb[(size_t)(kb + kr) * NHID + nb + c4];
      Bs[c4 + 0][kr] = f2bf(v.x);
      Bs[c4 + 1][kr] = f2bf(v.y);
      Bs[c4 + 2][kr] = f2bf(v.z);
      Bs[c4 + 3][kr] = f2bf(v.w);
    }
    __syncthreads();

    Frag a, bf[4];
    {
      const unsigned short* p = &As[waveM * 16 + r][0];
      a.q[0] = *(const uint4*)(p + kg * 16);
      a.q[1] = *(const uint4*)(p + kg * 16 + 8);
    }
    for (int j = 0; j < 4; j++) {
      const unsigned short* p = &Bs[waveN * 64 + j * 16 + r][kg * 16];
      bf[j].q[0] = *(const uint4*)(p);
      bf[j].q[1] = *(const uint4*)(p + 8);
    }
    for (int j = 0; j < 4; j++)
      acc[j] = __builtin_amdgcn_wmma_f32_16x16x32_bf16(
          false, a.v, false, bf[j].v, (short)0, acc[j], false, false);
    __syncthreads();
  }

  for (int j = 0; j < 4; j++) {
    int gcol = nb + waveN * 64 + j * 16 + r;
    int rowbase = waveM * 16 + kg * 8;
    for (int v = 0; v < 8; v++)
      vn[((size_t)b * NVIRT + rowbase + v) * NHID + gcol] = acc[j][v];
  }
}

// ---------------------------------------------------------------------------
// Graph / elementwise kernels
// ---------------------------------------------------------------------------
__global__ void init_kernel(float* __restrict__ agg, float* __restrict__ deg) {
  size_t i = (size_t)blockIdx.x * 256 + threadIdx.x;   // grid = TOTALN blocks
  agg[i] = 0.0f;
  if (i < TOTALN) deg[i] = 1.0f;                        // self-loop degree
}

__global__ void deg_kernel(const int* __restrict__ dst, float* __restrict__ deg) {
  int i = blockIdx.x * 256 + threadIdx.x;               // grid = NEDGE/256
  atomicAdd(&deg[dst[i]], 1.0f);
}

__global__ __launch_bounds__(64) void scatter_kernel(
    const int* __restrict__ src, const int* __restrict__ dst,
    const float* __restrict__ deg, const float* __restrict__ hw,
    float* __restrict__ agg)
{
  int e = blockIdx.x;
  int s = src[e], d = dst[e];
  float coef = rsqrtf(deg[s]) * rsqrtf(deg[d]);
  int t4 = threadIdx.x << 2;
  float4 hv = *(const float4*)&hw[(size_t)s * NHID + t4];
  float* ap = &agg[(size_t)d * NHID + t4];
  atomicAdd(ap + 0, hv.x * coef);
  atomicAdd(ap + 1, hv.y * coef);
  atomicAdd(ap + 2, hv.z * coef);
  atomicAdd(ap + 3, hv.w * coef);
}

__global__ void combine_kernel(const float* __restrict__ agg,
                               const float* __restrict__ hw,
                               const float* __restrict__ deg,
                               const float* __restrict__ bg,
                               float* __restrict__ g)
{
  size_t i = (size_t)blockIdx.x * 256 + threadIdx.x;    // grid = TOTALN blocks
  int node = (int)(i >> 8), h = (int)(i & 255);
  float val = agg[i] + hw[i] / deg[node] + bg[h];
  g[i] = fmaxf(val, 0.0f);
}

__global__ __launch_bounds__(256) void attn_kernel(const float* __restrict__ t,
                                                   float* __restrict__ att)
{
  int b = blockIdx.x, tid = threadIdx.x;
  __shared__ float proto[NHID];
  __shared__ float red[NHID];
  const float* tb = t + (size_t)b * NNODE * NHID;
  float s = 0.0f;
  for (int n = 0; n < NNODE; n++) s += tb[(size_t)n * NHID + tid];
  float p = s * (1.0f / NNODE);
  proto[tid] = p;
  red[tid] = p * p;
  __syncthreads();
  for (int off = 128; off > 0; off >>= 1) {
    if (tid < off) red[tid] += red[tid + off];
    __syncthreads();
  }
  float pn = fmaxf(sqrtf(red[0]), 1e-8f);
  int wave = tid >> 5, lane = tid & 31;
  for (int n = wave; n < NNODE; n += 8) {
    const float* tn = tb + (size_t)n * NHID;
    float dot = 0.0f, sq = 0.0f;
    for (int k = 0; k < 8; k++) {
      float v = tn[lane + k * 32];
      dot += v * proto[lane + k * 32];
      sq  += v * v;
    }
    for (int m = 16; m > 0; m >>= 1) {
      dot += __shfl_xor(dot, m, 32);
      sq  += __shfl_xor(sq,  m, 32);
    }
    if (lane == 0) {
      float tnorm = fmaxf(sqrtf(sq), 1e-8f);
      att[(size_t)b * NNODE + n] = 0.5f * (1.0f + dot / (tnorm * pn));
    }
  }
}

__global__ void scale_kernel(const float* __restrict__ ew,
                             const float* __restrict__ att,
                             float* __restrict__ scl)
{
  int i = blockIdx.x * 256 + threadIdx.x;               // grid = 128 blocks
  const float* e = ew + (size_t)i * NVIRT;
  float S = 0.0f;
  for (int v = 0; v < NVIRT; v++) S += e[v];
  float a = att[i];
  float rs = a * S;
  scl[i] = (rs == 0.0f) ? a : a / rs;
}

__global__ void mean_kernel(const float* __restrict__ vn2, float* __restrict__ gf) {
  int i = blockIdx.x * 256 + threadIdx.x;               // grid = 64 blocks -> B*H
  int b = i >> 8, h = i & 255;
  const float* v = vn2 + (size_t)b * NVIRT * NHID + h;
  float s = 0.0f;
  for (int k = 0; k < NVIRT; k++) s += v[(size_t)k * NHID];
  gf[i] = s * (1.0f / NVIRT);
}

__global__ __launch_bounds__(256) void head_kernel(
    const float* __restrict__ gf, const float* __restrict__ mW1,
    const float* __restrict__ mb1, const float* __restrict__ mW2,
    const float* __restrict__ mb2, float* __restrict__ out)
{
  int b = blockIdx.x, tid = threadIdx.x;
  __shared__ float gfs[NHID];
  __shared__ float hid[NHID];
  gfs[tid] = gf[(size_t)b * NHID + tid];
  __syncthreads();
  float s = mb1[tid];
  for (int k = 0; k < NHID; k++) s += gfs[k] * mW1[(size_t)k * NHID + tid];
  hid[tid] = fmaxf(s, 0.0f);
  __syncthreads();
  if (tid < 10) {
    float o = mb2[tid];
    for (int h = 0; h < NHID; h++) o += hid[h] * mW2[(size_t)h * 10 + tid];
    out[(size_t)b * 10 + tid] = o;
  }
}

// ---------------------------------------------------------------------------
extern "C" void kernel_launch(void* const* d_in, const int* in_sizes, int n_in,
                              void* d_out, int out_size, void* d_ws, size_t ws_size,
                              hipStream_t stream) {
  (void)in_sizes; (void)n_in; (void)out_size; (void)ws_size;
  const float* x     = (const float*)d_in[0];
  const int*   esrc  = (const int*)  d_in[1];
  const int*   edst  = (const int*)  d_in[2];
  const float* W_emb = (const float*)d_in[3];
  const float* b_emb = (const float*)d_in[4];
  const float* W_gcn = (const float*)d_in[5];
  const float* b_gcn = (const float*)d_in[6];
  const float* aW1   = (const float*)d_in[7];
  const float* ab1   = (const float*)d_in[8];
  const float* aW2   = (const float*)d_in[9];
  const float* ab2   = (const float*)d_in[10];
  const float* vW1   = (const float*)d_in[11];
  const float* vb1   = (const float*)d_in[12];
  const float* vW2   = (const float*)d_in[13];
  const float* vb2   = (const float*)d_in[14];
  const float* mW1   = (const float*)d_in[15];
  const float* mb1   = (const float*)d_in[16];
  const float* mW2   = (const float*)d_in[17];
  const float* mb2   = (const float*)d_in[18];
  const float* ew    = (const float*)d_in[19];
  float* out = (float*)d_out;

  const size_t BIG = (size_t)TOTALN * NHID;             // 8M floats = 32MB
  char* ws = (char*)d_ws;
  float* buf0 = (float*)ws;                             // h -> g
  float* buf1 = (float*)(ws + BIG * 4);                 // hw -> t
  float* buf2 = (float*)(ws + 2 * BIG * 4);             // agg -> t1 -> vn bufs
  float* deg  = (float*)(ws + 3 * BIG * 4);
  float* att  = deg + TOTALN;
  float* scl  = att + TOTALN;
  float* gf   = scl + TOTALN;
  float* vn0  = buf2;                                   // 64*64*256 = 1M floats
  float* vn1  = buf2 + 1048576;
  float* vn2  = buf2 + 2097152;

  dim3 blk(256);
  dim3 gBig(TOTALN / 128, NHID / 128);                  // (256, 2)
  dim3 gVN2(4096 / 128, NHID / 128);                    // (32, 2)

  // 1. h = x @ W_emb + b_emb
  gemm_bf16<<<gBig, blk, 0, stream>>>(x, W_emb, b_emb, buf0, TOTALN, 128, NHID, 0);
  // 2. hw = h @ W_gcn
  gemm_bf16<<<gBig, blk, 0, stream>>>(buf0, W_gcn, nullptr, buf1, TOTALN, NHID, NHID, 0);
  // 3. deg = 1 + scatter(ones), agg = 0
  init_kernel<<<TOTALN, blk, 0, stream>>>(buf2, deg);
  deg_kernel<<<NEDGE / 256, blk, 0, stream>>>(edst, deg);
  // 4. agg += coef * hw[src] scattered to dst
  scatter_kernel<<<NEDGE, dim3(64), 0, stream>>>(esrc, edst, deg, buf1, buf2);
  // 5. g = relu(agg + hw/deg + b_gcn)
  combine_kernel<<<TOTALN, blk, 0, stream>>>(buf2, buf1, deg, b_gcn, buf0);
  // 6. t1 = relu(g @ aW1 + ab1); t = t1 @ aW2 + ab2
  gemm_bf16<<<gBig, blk, 0, stream>>>(buf0, aW1, ab1, buf2, TOTALN, NHID, NHID, 1);
  gemm_bf16<<<gBig, blk, 0, stream>>>(buf2, aW2, ab2, buf1, TOTALN, NHID, NHID, 0);
  // 7. cosine attention vs graph prototype
  attn_kernel<<<NBAT, blk, 0, stream>>>(buf1, att);
  // 8. per-node normalization scale: scl = att/rs (or att if rs==0)
  scale_kernel<<<(NBAT * NNODE) / 256, blk, 0, stream>>>(ew, att, scl);
  // 9. vn = einsum('bnv,bnh->bvh', mw, g)
  vn_gemm<<<dim3(NBAT, NHID / 128), blk, 0, stream>>>(ew, scl, buf0, vn0);
  // 10. vn MLP
  gemm_bf16<<<gVN2, blk, 0, stream>>>(vn0, vW1, vb1, vn1, NBAT * NVIRT, NHID, NHID, 1);
  gemm_bf16<<<gVN2, blk, 0, stream>>>(vn1, vW2, vb2, vn2, NBAT * NVIRT, NHID, NHID, 0);
  // 11. gf = mean over V
  mean_kernel<<<NBAT, blk, 0, stream>>>(vn2, gf);
  // 12. out = relu(gf @ mW1 + mb1) @ mW2 + mb2
  head_kernel<<<NBAT, blk, 0, stream>>>(gf, mW1, mb1, mW2, mb2, out);
}
